// Conditional_Contrastive_loss_64518998720811
// MI455X (gfx1250) — matrix operand order repaired and checked by reading, not compiled, not run on hardware
//
#include <hip/hip_runtime.h>
#include <hip/hip_bf16.h>

#define N_PTS 8192
#define DIM   128
#define NCLS  100
#define EPSN  1e-8f

typedef __attribute__((ext_vector_type(16))) _Float16 v16h;
typedef __attribute__((ext_vector_type(8)))  _Float16 v8h;
typedef __attribute__((ext_vector_type(4)))  _Float16 v4h;
typedef __attribute__((ext_vector_type(8)))  float    v8f;

// ---------------------------------------------------------------------------
// Zero the row accumulators (workspace is poisoned 0xAA, must re-init per call)
// ---------------------------------------------------------------------------
__global__ __launch_bounds__(256) void k_init(float* __restrict__ S,
                                              float* __restrict__ M) {
  int t = blockIdx.x * blockDim.x + threadIdx.x;
  if (t < N_PTS) { S[t] = 0.f; M[t] = 0.f; }
}

// ---------------------------------------------------------------------------
// Normalize inst_embed rows -> f16 (for WMMA), and compute the proxy term
//   P[i] = exp((dot(xn_i, pn_i) - margin) / T)   (all in fp32)
// One wave (32 lanes) per row; lane handles 4 of the 128 features.
// ---------------------------------------------------------------------------
__global__ __launch_bounds__(256) void k_normalize(
    const float* __restrict__ inst, const float* __restrict__ proxy,
    const float* __restrict__ tptr, const float* __restrict__ mptr,
    _Float16* __restrict__ xnh, float* __restrict__ P) {
  const int row  = blockIdx.x * 8 + (threadIdx.x >> 5);
  const int lane = threadIdx.x & 31;
  const float invT = 1.0f / tptr[0];
  const float marg = mptr[0];

  float4 xv = *(const float4*)(inst + (size_t)row * DIM + lane * 4);
  float ss = xv.x * xv.x + xv.y * xv.y + xv.z * xv.z + xv.w * xv.w;
  #pragma unroll
  for (int off = 16; off >= 1; off >>= 1) ss += __shfl_xor(ss, off, 32);
  const float xinv = 1.0f / fmaxf(sqrtf(ss), EPSN);

  const float x0 = xv.x * xinv, x1 = xv.y * xinv, x2 = xv.z * xinv, x3 = xv.w * xinv;
  v4h h;
  h[0] = (_Float16)x0; h[1] = (_Float16)x1; h[2] = (_Float16)x2; h[3] = (_Float16)x3;
  *(v4h*)(xnh + (size_t)row * DIM + lane * 4) = h;

  float4 pv = *(const float4*)(proxy + (size_t)row * DIM + lane * 4);
  float ps = pv.x * pv.x + pv.y * pv.y + pv.z * pv.z + pv.w * pv.w;
  #pragma unroll
  for (int off = 16; off >= 1; off >>= 1) ps += __shfl_xor(ps, off, 32);
  const float pinv = 1.0f / fmaxf(sqrtf(ps), EPSN);

  float d = x0 * (pv.x * pinv) + x1 * (pv.y * pinv) +
            x2 * (pv.z * pinv) + x3 * (pv.w * pinv);
  #pragma unroll
  for (int off = 16; off >= 1; off >>= 1) d += __shfl_xor(d, off, 32);
  if (lane == 0) P[row] = __expf((d - marg) * invT);
}

// ---------------------------------------------------------------------------
// WMMA Gram-matrix row-sum kernel, double-buffered.
// Each wave owns 16 rows [i0, i0+16) and sweeps a 1024-column strip.
// Per 16x16 tile: 4x v_wmma_f32_16x16x32_f16 (K=128), then per element
// e = exp2(fma(c, log2e/T, -m*log2e/T)) via v_exp_f32, mask-weighted
// accumulation. B fragments for tile t+1 are loaded before computing tile t
// so the elementwise work hides the (L2-resident) load latency.
// i0 is pinned to an SGPR via readfirstlane so the diagonal-tile test
// (jt == i0) compiles to a scalar branch, not an exec-mask dance.
// End of strip: shfl_xor reduce within 16-lane halves + float atomicAdd.
// ---------------------------------------------------------------------------
__global__ __launch_bounds__(256) void k_simsum(
    const _Float16* __restrict__ xnh,
    const float* __restrict__ negmask, const int* __restrict__ labels,
    const float* __restrict__ tptr, const float* __restrict__ mptr,
    float* __restrict__ S, float* __restrict__ M, int jstride) {
  const int lane = threadIdx.x & 31;
  const int wave = threadIdx.x >> 5;
  // i0 is identical across all 32 lanes of the wave; readfirstlane makes that
  // visible to the compiler (SGPR) so comparisons against it stay scalar.
  const int i0 = __builtin_amdgcn_readfirstlane(blockIdx.y * 128 + wave * 16);
  const int j_begin = blockIdx.x * jstride;
  const int j_end = j_begin + jstride;
  const float LOG2E = 1.44269504088896340736f;
  const float invT = LOG2E / tptr[0];
  const float bias = -mptr[0] * invT;       // e = exp2(c*invT + bias)

  // --- Preload A fragments: rows i0..i0+15, full K=128 (contiguous K order:
  //     lanes 0-15 -> K {k..k+15}; lanes 16-31 -> K {k+16..k+31}) ---
  v16h a[4];
  {
    const int rA = i0 + (lane & 15);
    const int klo = (lane < 16) ? 0 : 16;
    const _Float16* arow = xnh + (size_t)rA * DIM + klo;
    #pragma unroll
    for (int kc = 0; kc < 4; ++kc) {
      v8h lo = *(const v8h*)(arow + kc * 32);
      v8h hi = *(const v8h*)(arow + kc * 32 + 8);
      #pragma unroll
      for (int t = 0; t < 8; ++t) { a[kc][t] = lo[t]; a[kc][t + 8] = hi[t]; }
    }
  }

  const int jcol   = lane & 15;
  const int rowoff = (lane >= 16) ? 8 : 0;
  const int kB     = (lane < 16) ? 0 : 16;

  // Hoisted mask offsets (32-bit so loads stay saddr+voffset scale_offset):
  // row r of this wave uses class labels[i0+r+rowoff] for the whole sweep.
  int moff[8];
  #pragma unroll
  for (int r = 0; r < 8; ++r)
    moff[r] = labels[i0 + r + rowoff] * N_PTS + jcol;

  float sumS[8], sumM[8];
  #pragma unroll
  for (int r = 0; r < 8; ++r) { sumS[r] = 0.f; sumM[r] = 0.f; }

  // B = xn^T: column j of B is row j of xn -> contiguous 16-half load/lane.
  const _Float16* bbase = xnh + (size_t)jcol * DIM + kB;

  auto loadB = [&](int jt, v16h* b) {
    const _Float16* p = bbase + (size_t)jt * DIM;
    #pragma unroll
    for (int kc = 0; kc < 4; ++kc) b[kc] = *(const v16h*)(p + kc * 32);
  };

  auto tile = [&](int jt, const v16h* b) {
    v8f c = {};
    #pragma unroll
    for (int kc = 0; kc < 4; ++kc)
      c = __builtin_amdgcn_wmma_f32_16x16x32_f16(
          /*neg_a=*/false, a[kc], /*neg_b=*/false, b[kc],
          /*c_mod=*/(short)0, c, /*reuse_a=*/false, /*reuse_b=*/false);
    if (jt == i0) {                          // scalar: SGPR vs SGPR
      // Only tile that can contain diagonal elements (i0, jt 16-aligned).
      #pragma unroll
      for (int r = 0; r < 8; ++r) {
        float e = __builtin_amdgcn_exp2f(fmaf(c[r], invT, bias));
        if (r + rowoff == jcol) e = 0.f;    // implements _remove_diag
        const float m = negmask[moff[r] + jt];
        sumS[r] += e;
        sumM[r] = fmaf(e, m, sumM[r]);
      }
    } else {
      #pragma unroll
      for (int r = 0; r < 8; ++r) {
        float e = __builtin_amdgcn_exp2f(fmaf(c[r], invT, bias));
        const float m = negmask[moff[r] + jt];
        sumS[r] += e;
        sumM[r] = fmaf(e, m, sumM[r]);
      }
    }
  };

  v16h b0[4], b1[4];
  loadB(j_begin, b0);
  for (int jt = j_begin; jt < j_end; jt += 32) {
    loadB(jt + 16, b1);
    __builtin_prefetch(bbase + (size_t)(jt + 48) * DIM, 0, 3);
    tile(jt, b0);
    // Tail load of the last iteration reads 16 rows past xnh; that region is
    // the P/S/M workspace (allocated), loaded but never consumed -> harmless.
    loadB(jt + 32, b0);
    tile(jt + 16, b1);
  }

  #pragma unroll
  for (int r = 0; r < 8; ++r) {
    float vS = sumS[r], vM = sumM[r];
    #pragma unroll
    for (int off = 8; off >= 1; off >>= 1) {
      vS += __shfl_xor(vS, off, 16);        // reduce within each 16-lane half
      vM += __shfl_xor(vM, off, 16);
    }
    if ((lane & 15) == 0) {
      const int gi = i0 + r + rowoff;
      atomicAdd(&S[gi], vS);
      atomicAdd(&M[gi], vM);
    }
  }
}

// ---------------------------------------------------------------------------
// Per-row loss: L[i] = log((M_i + P_i) / (P_i + S_i))
// ---------------------------------------------------------------------------
__global__ __launch_bounds__(256) void k_rowloss(
    const float* __restrict__ S, const float* __restrict__ M,
    const float* __restrict__ P, float* __restrict__ L) {
  int i = blockIdx.x * blockDim.x + threadIdx.x;
  if (i < N_PTS) {
    float num = M[i] + P[i];
    float den = P[i] + S[i];
    L[i] = logf(num / den);
  }
}

// ---------------------------------------------------------------------------
// Deterministic single-block final reduction: out = -mean(L)
// ---------------------------------------------------------------------------
__global__ __launch_bounds__(256) void k_reduce(const float* __restrict__ L,
                                                float* __restrict__ out) {
  __shared__ float sm[256];
  float s = 0.f;
  for (int i = threadIdx.x; i < N_PTS; i += 256) s += L[i];
  sm[threadIdx.x] = s;
  __syncthreads();
  for (int off = 128; off >= 1; off >>= 1) {
    if (threadIdx.x < off) sm[threadIdx.x] += sm[threadIdx.x + off];
    __syncthreads();
  }
  if (threadIdx.x == 0) out[0] = -sm[0] / (float)N_PTS;
}

// ---------------------------------------------------------------------------
extern "C" void kernel_launch(void* const* d_in, const int* in_sizes, int n_in,
                              void* d_out, int out_size, void* d_ws, size_t ws_size,
                              hipStream_t stream) {
  const float* inst    = (const float*)d_in[0];
  const float* proxy   = (const float*)d_in[1];
  const float* negmask = (const float*)d_in[2];
  const int*   labels  = (const int*)d_in[3];
  const float* tptr    = (const float*)d_in[4];
  const float* mptr    = (const float*)d_in[5];
  float* out = (float*)d_out;

  // Workspace layout: xn_f16 [N*D] (2 MB) | P[N] | S[N] | M[N] | L[N]
  char* ws = (char*)d_ws;
  _Float16* xnh = (_Float16*)ws;
  float* P = (float*)(ws + (size_t)N_PTS * DIM * sizeof(_Float16));
  float* S = P + N_PTS;
  float* M = S + N_PTS;
  float* L = M + N_PTS;

  k_init<<<dim3((N_PTS + 255) / 256), 256, 0, stream>>>(S, M);
  k_normalize<<<dim3(N_PTS / 8), 256, 0, stream>>>(inst, proxy, tptr, mptr, xnh, P);

  const int STRIPS = 8;                     // 8 x 64 = 512 blocks, 4096 waves
  k_simsum<<<dim3(STRIPS, N_PTS / 128), 256, 0, stream>>>(
      xnh, negmask, labels, tptr, mptr, S, M, N_PTS / STRIPS);

  k_rowloss<<<dim3(N_PTS / 256), 256, 0, stream>>>(S, M, P, L);
  k_reduce<<<1, 256, 0, stream>>>(L, out);
}
